// AWFLN_25958782337147
// MI455X (gfx1250) — matrix operand order repaired and checked by reading, not compiled, not run on hardware
//
#include <hip/hip_runtime.h>

typedef __attribute__((ext_vector_type(16))) _Float16 v16h;
typedef __attribute__((ext_vector_type(8)))  float    v8f;

#define ACT_NONE  0
#define ACT_PRELU 1
#define ACT_RELU  2

// ---------------------------------------------------------------------------
// Weight pre-pack: convert OIHW f32 weights into ready-to-load WMMA B
// fragments (f16), tap-major K order, zero-padded in both K (to 32) and N
// (to 32 = 2 tiles per wave).  Layout: [tile][chunk][lane][e] halves.
//   kloc(lane,e) = (e<8 ? e : e+8) + 8*(lane>=16)   (16-bit B 32x16 layout)
//   n            = tile*16 + (lane&15)
//   chunk q      = p*cChunks + ci  ->  reduction k = channel (ci*32 + kloc), tap p
// ---------------------------------------------------------------------------
__global__ void awfln_pack_w(const float* __restrict__ wgt, _Float16* __restrict__ out,
                             int Cout, int Cin, int cChunks, int ntilesP)
{
    const int nchunks = 9 * cChunks;
    const int total   = ntilesP * nchunks * 512;
    int idx = blockIdx.x * blockDim.x + threadIdx.x;
    if (idx >= total) return;
    const int e    = idx & 15;
    const int lane = (idx >> 4) & 31;
    const int rest = idx >> 9;
    const int q    = rest % nchunks;
    const int t    = rest / nchunks;
    const int p    = q / cChunks;
    const int ci   = q - p * cChunks;
    const int kloc = ((e < 8) ? e : e + 8) + ((lane >> 4) & 1) * 8;
    const int c    = ci * 32 + kloc;
    const int n    = t * 16 + (lane & 15);
    float v = 0.0f;
    if (c < Cin && n < Cout) v = wgt[(n * Cin + c) * 9 + p];
    out[idx] = (_Float16)v;
}

// ---------------------------------------------------------------------------
// Implicit-GEMM 3x3 conv (pad = dil, stride 1) via WMMA f32<-f16.
// One wave per block: 16 pixels (one row segment) x 32 out-channels
// (two 16x16 N tiles sharing one A fragment).  K runs tap-major so the
// attention factor (AFLB einsum mode) is a single scalar per chunk.
// ---------------------------------------------------------------------------
template<int H, int W, bool HAS_ATT>
__global__ __launch_bounds__(32)
void awfln_conv3x3_wmma(const float* __restrict__ in,
                        const _Float16* __restrict__ pw,   // packed weights
                        const float* __restrict__ bias,
                        const float* __restrict__ alpha,
                        const float* __restrict__ att,     // [B,9,H,W] or null
                        const float* __restrict__ res,     // residual or null
                        float* __restrict__ out,           // [B,CtotOut,H,W]
                        int Cin, int Cout, int CtotOut, int outChOff,
                        int dil, int act, int cChunks)
{
    constexpr int HW = H * W;
    const int lane = threadIdx.x;                 // full wave, EXEC all ones
    const int m0   = blockIdx.x * 16;             // 16 pixels, same row
    const int tp   = blockIdx.y * 2;              // N-tile pair
    const int x0   = m0 % W;                      // uniform
    const int y0   = (m0 / W) % H;                // uniform
    const int b    = m0 / HW;                     // uniform
    const int ml   = lane & 15;
    const int xA   = x0 + ml;                     // A-row pixel (M = lane&15)
    const int hiA  = (lane >= 16) ? 8 : 0;        // A layout: lanes 16-31 hold K+8/K+24
    const int nchunks = 9 * cChunks;

    const _Float16* pwl0 = pw + ((size_t)tp       * nchunks * 32 + lane) * 16;
    const _Float16* pwl1 = pw + ((size_t)(tp + 1) * nchunks * 32 + lane) * 16;

    v8f acc0 = {}, acc1 = {};

    #pragma unroll
    for (int p = 0; p < 9; ++p) {
        const int dy   = p / 3 - 1, dx = p % 3 - 1;
        const int yy   = y0 + dy * dil;                    // uniform
        const int xoff = dx * dil;
        const int xx   = xA + xoff;                        // per lane
        const bool okY  = (unsigned)yy < (unsigned)H;      // uniform
        const bool okXu = (x0 + xoff >= 0) && (x0 + 15 + xoff < W);  // uniform
        const bool okL  = okY && ((unsigned)xx < (unsigned)W);       // per lane

        float attv = 1.0f;
        if (HAS_ATT) attv = att[((b * 9 + p) * H + y0) * W + xA];

        const float* srcRow = in + ((b * Cin) * H + yy) * W + xx;    // valid iff okL

        for (int ci = 0; ci < cChunks; ++ci) {
            const int c0 = ci * 32;
            const int q  = p * cChunks + ci;

            const v16h bf0 = *(const v16h*)(pwl0 + (size_t)q * 512);
            const v16h bf1 = *(const v16h*)(pwl1 + (size_t)q * 512);

            float vals[16];
            if (okY && okXu && (c0 + 32 <= Cin)) {
                // interior fast path: 16 unconditional loads, one wait
                #pragma unroll
                for (int e = 0; e < 16; ++e) {
                    const int ke = ((e < 8) ? e : e + 8) + hiA;
                    vals[e] = srcRow[(c0 + ke) * HW];
                }
            } else {
                #pragma unroll
                for (int e = 0; e < 16; ++e) {
                    const int ke = ((e < 8) ? e : e + 8) + hiA;
                    const int c  = c0 + ke;
                    const bool g = okL && (c < Cin);
                    vals[e] = g ? srcRow[c * HW] : 0.0f;
                }
            }

            v16h afrag;
            #pragma unroll
            for (int e = 0; e < 16; ++e)
                afrag[e] = (_Float16)(HAS_ATT ? vals[e] * attv : vals[e]);

            acc0 = __builtin_amdgcn_wmma_f32_16x16x32_f16(
                false, afrag, false, bf0, (short)0, acc0, false, false);
            acc1 = __builtin_amdgcn_wmma_f32_16x16x32_f16(
                false, afrag, false, bf1, (short)0, acc1, false, false);
        }
    }

    // ---- epilogue: C/D layout  n = tile*16 + (lane&15),  m = m0 + v + 8*(lane>=16)
    const float al = alpha ? alpha[0] : 0.0f;
    #pragma unroll
    for (int t = 0; t < 2; ++t) {
        const int n = (tp + t) * 16 + ml;
        if (n >= Cout) continue;
        const float bv = bias ? bias[n] : 0.0f;
        const v8f& acc = t ? acc1 : acc0;
        #pragma unroll
        for (int v = 0; v < 8; ++v) {
            const int m = m0 + v + hiA;
            const int x = m % W;
            const int y = (m / W) % H;
            const int bb = m / HW;
            float r = acc[v] + bv;
            const int o = ((bb * CtotOut + outChOff + n) * H + y) * W + x;
            if (res) r += res[o];
            if (act == ACT_PRELU)     r = (r >= 0.0f) ? r : al * r;
            else if (act == ACT_RELU) r = (r >= 0.0f) ? r : 0.0f;
            out[o] = r;
        }
    }
}

// pixel_shuffle(r=4) of up-conv + PReLU into data1[:,0:4], pan into data1[:,4]
__global__ void awfln_shuffle_prelu_concat(const float* __restrict__ convup,
                                           const float* __restrict__ pan,
                                           const float* __restrict__ alpha,
                                           float* __restrict__ out, int total)
{
    int i = blockIdx.x * blockDim.x + threadIdx.x;
    if (i >= total) return;
    const int x = i & 255;
    const int y = (i >> 8) & 255;
    const int c = (i >> 16) % 5;
    const int b = i / (5 << 16);
    if (c == 4) {
        out[i] = pan[(b << 16) + (y << 8) + x];
    } else {
        const int h = y >> 2, r1 = y & 3, w = x >> 2, r2 = x & 3;
        const float v = convup[(((b * 64) + (c * 16 + r1 * 4 + r2)) * 64 + h) * 64 + w];
        const float a = alpha[0];
        out[i] = (v >= 0.0f) ? v : a * v;
    }
}

// in-place softmax over 9 attention channels, one thread per (b,y,x)
__global__ void awfln_softmax9(float* __restrict__ att, int B, int HW)
{
    int q = blockIdx.x * blockDim.x + threadIdx.x;
    if (q >= B * HW) return;
    const int b = q / HW;
    const int s = q - b * HW;
    float* p = att + (size_t)b * 9 * HW + s;
    float v[9];
    float mx = -1e30f;
    #pragma unroll
    for (int i = 0; i < 9; ++i) { v[i] = p[(size_t)i * HW]; mx = fmaxf(mx, v[i]); }
    float sum = 0.0f;
    #pragma unroll
    for (int i = 0; i < 9; ++i) { v[i] = expf(v[i] - mx); sum += v[i]; }
    const float inv = 1.0f / sum;
    #pragma unroll
    for (int i = 0; i < 9; ++i) p[(size_t)i * HW] = v[i] * inv;
}

extern "C" void kernel_launch(void* const* d_in, const int* in_sizes, int n_in,
                              void* d_out, int out_size, void* d_ws, size_t ws_size,
                              hipStream_t stream)
{
    (void)in_sizes; (void)n_in; (void)out_size; (void)ws_size;

    const float* ms_up   = (const float*)d_in[0];
    const float* ms_org  = (const float*)d_in[1];
    const float* pan     = (const float*)d_in[2];
    const float* w_up4   = (const float*)d_in[3];
    const float* b_up4   = (const float*)d_in[4];
    const float* a_up4   = (const float*)d_in[5];
    const float* w_blk53 = (const float*)d_in[6];
    const float* b_blk53 = (const float*)d_in[7];
    const float* a_blk53 = (const float*)d_in[8];
    const float* w_blk43 = (const float*)d_in[9];
    const float* b_blk43 = (const float*)d_in[10];
    const float* a_blk43 = (const float*)d_in[11];
    const float* w1a = (const float*)d_in[12]; const float* b1a = (const float*)d_in[13];
    const float* w1b = (const float*)d_in[14]; const float* b1b = (const float*)d_in[15];
    const float* w1c = (const float*)d_in[16]; const float* b1c = (const float*)d_in[17];
    const float* aw1 = (const float*)d_in[18]; const float* ab1 = (const float*)d_in[19];
    const float* fw1 = (const float*)d_in[20];
    const float* aw2 = (const float*)d_in[21]; const float* ab2 = (const float*)d_in[22];
    const float* fw2 = (const float*)d_in[23];
    const float* w_blk2 = (const float*)d_in[24]; const float* b_blk2 = (const float*)d_in[25];
    const float* a_blk2 = (const float*)d_in[26];
    const float* w2a = (const float*)d_in[27]; const float* b2a = (const float*)d_in[28];
    const float* w2b = (const float*)d_in[29]; const float* b2b = (const float*)d_in[30];
    const float* w2c = (const float*)d_in[31]; const float* b2c = (const float*)d_in[32];
    const float* aw3 = (const float*)d_in[33]; const float* ab3 = (const float*)d_in[34];
    const float* fw3 = (const float*)d_in[35];
    const float* aw4 = (const float*)d_in[36]; const float* ab4 = (const float*)d_in[37];
    const float* fw4 = (const float*)d_in[38];
    const float* w_c6 = (const float*)d_in[39]; const float* b_c6 = (const float*)d_in[40];

    float* out = (float*)d_out;

    // ---- workspace layout: float buffers, then packed-weight (f16) region
    const size_t S   = 256 * 256;
    const size_t S60 = (size_t)2 * 60 * S;
    float* ws  = (float*)d_ws;
    float* tUp = ws;                                   // [2,64,64,64]
    float* tD1 = tUp + (size_t)2 * 64 * 64 * 64;       // [2,5,256,256]
    float* C   = tD1 + (size_t)2 * 5 * S;              // [2,60,H,W] (reused as G)
    float* D   = C + S60;                              // (reused as H)
    float* E   = D + S60;                              // (reused as I)
    float* F   = E + S60;                              // (reused as J)
    float* ATT = F + S60;                              // [2,9,H,W]
    _Float16* packBase = (_Float16*)(ATT + (size_t)2 * 9 * S);
    size_t packCur = 0;
    float* G = C; float* Hh = D; float* I = E; float* J = F;

    // pack a weight tensor; returns its packed pointer
    auto pack = [&](const float* w, int Cout, int Cin) -> _Float16* {
        const int cChunks = (Cin + 31) / 32;
        const int ntilesP = 2 * ((Cout + 31) / 32);
        const int total   = ntilesP * 9 * cChunks * 512;
        _Float16* dst = packBase + packCur;
        packCur += (size_t)total;
        awfln_pack_w<<<(total + 255) / 256, 256, 0, stream>>>(w, dst, Cout, Cin,
                                                              cChunks, ntilesP);
        return dst;
    };

    _Float16* p_up4   = pack(w_up4,   64, 4);
    _Float16* p_blk43 = pack(w_blk43, 30, 4);
    _Float16* p_blk53 = pack(w_blk53, 30, 5);
    _Float16* p_w1a = pack(w1a, 20, 60);
    _Float16* p_w1b = pack(w1b, 20, 60);
    _Float16* p_w1c = pack(w1c, 20, 60);
    _Float16* p_aw1 = pack(aw1,  9, 60);
    _Float16* p_fw1 = pack(fw1, 60, 60);
    _Float16* p_aw2 = pack(aw2,  9, 60);
    _Float16* p_fw2 = pack(fw2, 60, 60);
    _Float16* p_blk2 = pack(w_blk2, 30, 60);
    _Float16* p_w2a = pack(w2a, 10, 30);
    _Float16* p_w2b = pack(w2b, 10, 30);
    _Float16* p_w2c = pack(w2c, 10, 30);
    _Float16* p_aw3 = pack(aw3,  9, 30);
    _Float16* p_fw3 = pack(fw3, 30, 30);
    _Float16* p_aw4 = pack(aw4,  9, 30);
    _Float16* p_fw4 = pack(fw4, 30, 30);
    _Float16* p_c6  = pack(w_c6,  4, 30);

    // conv launchers (H=W=256 with/without attention, and 64x64 up-conv)
    auto conv = [&](const float* in, const _Float16* pw, const float* b,
                    const float* a, const float* res, float* o,
                    int Cin, int Cout, int Ctot, int off, int dil, int act) {
        dim3 grid((unsigned)(2 * S / 16), (unsigned)((Cout + 31) / 32));
        awfln_conv3x3_wmma<256, 256, false><<<grid, 32, 0, stream>>>(
            in, pw, b, a, nullptr, res, o, Cin, Cout, Ctot, off, dil, act,
            (Cin + 31) / 32);
    };
    auto convAtt = [&](const float* in, const _Float16* pw, const float* res,
                       float* o, int Cin, int Cout, int act) {
        dim3 grid((unsigned)(2 * S / 16), (unsigned)((Cout + 31) / 32));
        awfln_conv3x3_wmma<256, 256, true><<<grid, 32, 0, stream>>>(
            in, pw, nullptr, nullptr, ATT, res, o, Cin, Cout, Cout, 0, 1, act,
            (Cin + 31) / 32);
    };
    auto softmax = [&]() {
        int n = 2 * (int)S;
        awfln_softmax9<<<(n + 255) / 256, 256, 0, stream>>>(ATT, 2, (int)S);
    };

    // 1) up-conv (64x64) -> pixel_shuffle(4)+PReLU, concat pan -> data1
    {
        dim3 grid((unsigned)(2 * 64 * 64 / 16), 2u);
        awfln_conv3x3_wmma<64, 64, false><<<grid, 32, 0, stream>>>(
            ms_org, p_up4, b_up4, nullptr, nullptr, nullptr, tUp,
            4, 64, 64, 0, 1, ACT_NONE, 1);
        int n = 2 * 5 * (int)S;
        awfln_shuffle_prelu_concat<<<(n + 255) / 256, 256, 0, stream>>>(tUp, pan, a_up4, tD1, n);
    }

    // 2) out1 = concat(prelu(conv(ms_up)), prelu(conv(data1))) -> C [2,60,H,W]
    conv(ms_up, p_blk43, b_blk43, a_blk43, nullptr, C, 4, 30, 60, 0, 1, ACT_PRELU);
    conv(tD1,   p_blk53, b_blk53, a_blk53, nullptr, C, 5, 30, 60, 30, 1, ACT_PRELU);

    // 3) rmrs (60ch): dilated branches -> D, AFLB x2, relu(+C) -> F
    conv(C, p_w1a, b1a, nullptr, nullptr, D, 60, 20, 60, 0, 1, ACT_NONE);
    conv(C, p_w1b, b1b, nullptr, nullptr, D, 60, 20, 60, 20, 2, ACT_NONE);
    conv(C, p_w1c, b1c, nullptr, nullptr, D, 60, 20, 60, 40, 3, ACT_NONE);
    conv(D, p_aw1, ab1, nullptr, nullptr, ATT, 60, 9, 9, 0, 1, ACT_NONE);
    softmax();
    convAtt(D, p_fw1, nullptr, E, 60, 60, ACT_NONE);
    conv(E, p_aw2, ab2, nullptr, nullptr, ATT, 60, 9, 9, 0, 1, ACT_NONE);
    softmax();
    convAtt(E, p_fw2, C, F, 60, 60, ACT_RELU);

    // 4) blk2: prelu(conv(out2)) -> G [2,30,H,W]
    conv(F, p_blk2, b_blk2, a_blk2, nullptr, G, 60, 30, 30, 0, 1, ACT_PRELU);

    // 5) rmrs (30ch): branches -> Hh, AFLB x2, relu(+G) -> J
    conv(G, p_w2a, b2a, nullptr, nullptr, Hh, 30, 10, 30, 0, 1, ACT_NONE);
    conv(G, p_w2b, b2b, nullptr, nullptr, Hh, 30, 10, 30, 10, 2, ACT_NONE);
    conv(G, p_w2c, b2c, nullptr, nullptr, Hh, 30, 10, 30, 20, 3, ACT_NONE);
    conv(Hh, p_aw3, ab3, nullptr, nullptr, ATT, 30, 9, 9, 0, 1, ACT_NONE);
    softmax();
    convAtt(Hh, p_fw3, nullptr, I, 30, 30, ACT_NONE);
    conv(I, p_aw4, ab4, nullptr, nullptr, ATT, 30, 9, 9, 0, 1, ACT_NONE);
    softmax();
    convAtt(I, p_fw4, G, J, 30, 30, ACT_RELU);

    // 6) final conv + ms_up residual -> d_out
    conv(J, p_c6, b_c6, nullptr, ms_up, out, 30, 4, 4, 0, 1, ACT_NONE);
}
